// BertSelfAttention_17471926960865
// MI455X (gfx1250) — matrix operand order repaired and checked by reading, not compiled, not run on hardware
//
#include <hip/hip_runtime.h>
#include <hip/hip_bf16.h>

#define BB 4
#define SS 2048
#define HH 16
#define DH 64
#define DM 1024

typedef __bf16 bf16;
typedef __attribute__((ext_vector_type(16))) __bf16 v16bf;
typedef __attribute__((ext_vector_type(8)))  float  v8f;
typedef __attribute__((ext_vector_type(4)))  unsigned int uv4;
typedef __attribute__((ext_vector_type(4)))  unsigned int u32x4;
typedef __attribute__((ext_vector_type(8)))  int i32x8;
typedef __attribute__((ext_vector_type(4)))  int i32x4;

union FragU { uv4 u[2]; v16bf v; };
union V8U  { v8f v; float f[8]; };

// ---- fragment loaders (bf16, 16x16x32 WMMA) ----
// A (16xK=32): lane = M (lane&15); per-lane halves: K = kh..kh+7, kh+16..kh+23, kh = (lane>>4)*8
__device__ __forceinline__ v16bf load_fragA(const bf16* base, int ld) {
    const int lane = threadIdx.x & 31;
    const bf16* p = base + (lane & 15) * ld + ((lane >> 4) << 3);
    FragU f;
    f.u[0] = *reinterpret_cast<const uv4*>(p);
    f.u[1] = *reinterpret_cast<const uv4*>(p + 16);
    return f.v;
}
// B (K=32 x 16): lane = N (lane&15); per-lane halves: K = kh..kh+7, kh+16..kh+23, kh = (lane>>4)*8
__device__ __forceinline__ v16bf load_fragB(const bf16* base, int ld) {
    const int lane = threadIdx.x & 31;
    const bf16* p = base + (lane & 15) * ld + ((lane >> 4) << 3);
    FragU f;
    f.u[0] = *reinterpret_cast<const uv4*>(p);
    f.u[1] = *reinterpret_cast<const uv4*>(p + 16);
    return f.v;
}

__device__ __forceinline__ v8f wmma_bf16(v16bf a, v16bf b, v8f c) {
    return __builtin_amdgcn_wmma_f32_16x16x32_bf16(false, a, false, b, (short)0, c, false, false);
}

__device__ __forceinline__ float rmax16(float x) {
    #pragma unroll
    for (int off = 1; off < 16; off <<= 1) x = fmaxf(x, __shfl_xor(x, off, 16));
    return x;
}
__device__ __forceinline__ float rsum16(float x) {
    #pragma unroll
    for (int off = 1; off < 16; off <<= 1) x += __shfl_xor(x, off, 16);
    return x;
}

// ---- Tensor Data Mover: 2D bf16 tile (rows x cols), global row stride ldg (elems),
// DMA'd into LDS at lptr, packed row-major with ld = cols. ----
__device__ __forceinline__ unsigned lds_off(const bf16* p) {
    return (unsigned)(size_t)(const __attribute__((address_space(3))) bf16*)p;
}

__device__ __forceinline__ void tdm_load_tile(const bf16* gptr, const bf16* lptr,
                                              int cols, int rows, int ldg) {
    size_t ga = (size_t)gptr;
    u32x4 g0;
    g0[0] = 1u;                                             // count=1, user mode
    g0[1] = lds_off(lptr);                                  // lds_addr (bits 63:32)
    g0[2] = (unsigned)(ga & 0xFFFFFFFFull);                 // global_addr lo
    g0[3] = (unsigned)((ga >> 32) & 0x1FFFFFFull)           // global_addr[56:32]
          | (2u << 30);                                     // type = 2 ("image")
    i32x8 g1;
    g1[0] = (1 << 16);                                      // wg_mask=0, data_size=1 (2 bytes)
    g1[1] = (ldg & 0xFFFF) << 16;                           // tensor_dim0[15:0] = ldg
    g1[2] = (int)(((unsigned)ldg) >> 16)                    // tensor_dim0[31:16]
          | ((8192 & 0xFFFF) << 16);                        // tensor_dim1[15:0] = 8192
    g1[3] = (8192 >> 16) | (cols << 16);                    // tensor_dim1 hi | tile_dim0
    g1[4] = rows;                                           // tile_dim1 | tile_dim2=0
    g1[5] = ldg;                                            // tensor_dim0_stride lo
    g1[6] = 0;                                              // stride0 hi | stride1 lo
    g1[7] = 0;                                              // stride1 hi
    i32x4 z4 = {0, 0, 0, 0};
    i32x8 z8 = {0, 0, 0, 0, 0, 0, 0, 0};
    __builtin_amdgcn_tensor_load_to_lds(g0, g1, z4, z4, z8, 0);
}

// ---- kernel 1: X fp32 -> bf16 ----
__global__ void cvt_x_kernel(const float* __restrict__ x, bf16* __restrict__ y) {
    size_t i = ((size_t)blockIdx.x * blockDim.x + threadIdx.x) * 4;
    float4 v = *reinterpret_cast<const float4*>(x + i);
    y[i + 0] = (bf16)v.x; y[i + 1] = (bf16)v.y;
    y[i + 2] = (bf16)v.z; y[i + 3] = (bf16)v.w;
}

// ---- kernel 2: Wt[mat][n][k] = W_mat[k][n], fp32 -> bf16 ----
__global__ void wt_kernel(const float* __restrict__ Wq, const float* __restrict__ Wk,
                          const float* __restrict__ Wv, bf16* __restrict__ Wt) {
    const float* W = (blockIdx.z == 0) ? Wq : ((blockIdx.z == 1) ? Wk : Wv);
    int idx = blockIdx.x * 256 + threadIdx.x;   // over DM*DM
    int k = idx >> 10, n = idx & (DM - 1);
    Wt[(size_t)blockIdx.z * DM * DM + (size_t)n * DM + k] = (bf16)W[idx];
}

// ---- kernel 3: QKV projection GEMM, TDM double-buffered LDS staging ----
// grid (128, 16, 3), block 128 (4 waves). Block tile 64x64, K-step 64.
__global__ void __launch_bounds__(128) qkv_gemm_kernel(
    const bf16* __restrict__ X, const bf16* __restrict__ Wt,
    const float* __restrict__ bq, const float* __restrict__ bk, const float* __restrict__ bv,
    bf16* __restrict__ Qo, bf16* __restrict__ Ko, bf16* __restrict__ Vto) {
    const int mat  = blockIdx.z;
    const int w    = threadIdx.x >> 5;
    const int lane = threadIdx.x & 31;
    const int m0 = blockIdx.x * 64;
    const int n0 = blockIdx.y * 64;
    const int am = (w >> 1) * 32;   // wave sub-tile offsets inside block tile
    const int an = (w & 1) * 32;
    const bf16*  Wm   = Wt + (size_t)mat * DM * DM;
    const bf16*  Ag   = X  + (size_t)m0 * DM;
    const bf16*  Bg   = Wm + (size_t)n0 * DM;
    const float* bias = (mat == 0) ? bq : ((mat == 1) ? bk : bv);

    __shared__ __align__(16) bf16 Abuf[2][64 * 64];
    __shared__ __align__(16) bf16 Bbuf[2][64 * 64];

    V8U acc[2][2];
    #pragma unroll
    for (int i = 0; i < 2; i++)
        #pragma unroll
        for (int j = 0; j < 2; j++) { v8f z = {}; acc[i][j].v = z; }

    if (w == 0) {
        tdm_load_tile(Ag, Abuf[0], 64, 64, DM);
        tdm_load_tile(Bg, Bbuf[0], 64, 64, DM);
    }
    const int NSTEP = DM / 64;
    for (int kk = 0; kk < NSTEP; ++kk) {
        const int cur = kk & 1;
        if (w == 0) __builtin_amdgcn_s_wait_tensorcnt(0);
        __syncthreads();
        if (w == 0 && kk + 1 < NSTEP) {
            tdm_load_tile(Ag + (kk + 1) * 64, Abuf[cur ^ 1], 64, 64, DM);
            tdm_load_tile(Bg + (kk + 1) * 64, Bbuf[cur ^ 1], 64, 64, DM);
        }
        #pragma unroll
        for (int kc = 0; kc < 64; kc += 32) {
            v16bf a0 = load_fragA(&Abuf[cur][am * 64] + kc, 64);
            v16bf a1 = load_fragA(&Abuf[cur][(am + 16) * 64] + kc, 64);
            v16bf b0 = load_fragB(&Bbuf[cur][an * 64] + kc, 64);
            v16bf b1 = load_fragB(&Bbuf[cur][(an + 16) * 64] + kc, 64);
            acc[0][0].v = wmma_bf16(a0, b0, acc[0][0].v);
            acc[0][1].v = wmma_bf16(a0, b1, acc[0][1].v);
            acc[1][0].v = wmma_bf16(a1, b0, acc[1][0].v);
            acc[1][1].v = wmma_bf16(a1, b1, acc[1][1].v);
        }
    }

    bf16* dst = (mat == 0) ? Qo : ((mat == 1) ? Ko : Vto);
    #pragma unroll
    for (int i = 0; i < 2; i++) {
        #pragma unroll
        for (int j = 0; j < 2; j++) {
            int n = n0 + an + 16 * j + (lane & 15);
            float bias_n = bias[n];
            int h = n >> 6, d = n & 63;
            #pragma unroll
            for (int r = 0; r < 8; r++) {
                int m = m0 + am + 16 * i + r + 8 * (lane >> 4);
                int b = m >> 11, s = m & (SS - 1);
                float val = acc[i][j].f[r] + bias_n;
                if (mat < 2)  // Q, K: [b,h,s,d]
                    dst[(((size_t)(b * HH + h)) * SS + s) * DH + d] = (bf16)val;
                else          // V transposed: [b,h,d,s]
                    dst[(((size_t)(b * HH + h)) * DH + d) * SS + s] = (bf16)val;
            }
        }
    }
}

// ---- kernel 4: flash attention with TDM-staged K/V tiles ----
// grid (S/64, H, B), block 128 (4 waves x 16 q-rows)
__global__ void __launch_bounds__(128) attn_kernel(
    const bf16* __restrict__ Q, const bf16* __restrict__ K, const bf16* __restrict__ Vt,
    const float* __restrict__ mask, float* __restrict__ out) {
    const int qt = blockIdx.x, h = blockIdx.y, b = blockIdx.z;
    const int w = threadIdx.x >> 5, lane = threadIdx.x & 31;
    const int q0 = qt * 64 + w * 16;
    const bf16* Qh = Q  + ((size_t)(b * HH + h)) * SS * DH;
    const bf16* Kh = K  + ((size_t)(b * HH + h)) * SS * DH;
    const bf16* Vh = Vt + ((size_t)(b * HH + h)) * DH * SS;
    const float* mb = mask + (size_t)b * SS;

    __shared__ __align__(16) bf16 Kbuf[2][64 * 64];   // [key][d]
    __shared__ __align__(16) bf16 Vbuf[2][64 * 64];   // [d][key]
    __shared__ __align__(16) bf16 pbuf[4][16 * 64];   // per-wave P tile
    bf16* pw = &pbuf[w][0];

    v16bf qa0 = load_fragA(Qh + (size_t)q0 * DH, DH);       // d 0..31
    v16bf qa1 = load_fragA(Qh + (size_t)q0 * DH + 32, DH);  // d 32..63

    float mrow[8], lrow[8];
    V8U acc[4];
    #pragma unroll
    for (int r = 0; r < 8; r++) { mrow[r] = -1e30f; lrow[r] = 0.0f; }
    #pragma unroll
    for (int t = 0; t < 4; t++) { v8f z = {}; acc[t].v = z; }

    const float scale = 0.125f;  // 1/sqrt(64)

    if (w == 0) {
        tdm_load_tile(Kh, Kbuf[0], 64, 64, DH);  // keys 0..63, all d
        tdm_load_tile(Vh, Vbuf[0], 64, 64, SS);  // d 0..63, keys 0..63
    }

    for (int kt = 0; kt < SS / 64; ++kt) {
        const int k0  = kt * 64;
        const int cur = kt & 1;
        if (w == 0) __builtin_amdgcn_s_wait_tensorcnt(0);
        __syncthreads();
        if (w == 0 && kt + 1 < SS / 64) {
            tdm_load_tile(Kh + (size_t)(k0 + 64) * DH, Kbuf[cur ^ 1], 64, 64, DH);
            tdm_load_tile(Vh + (k0 + 64),              Vbuf[cur ^ 1], 64, 64, SS);
        }

        // ---- S = Q K^T over 64 keys (4 n-tiles x 2 d-chunks) ----
        V8U sc[4];
        #pragma unroll
        for (int j = 0; j < 4; j++) {
            v8f z = {};
            v16bf kb0 = load_fragB(&Kbuf[cur][(j * 16) * 64], 64);       // d 0..31
            v16bf kb1 = load_fragB(&Kbuf[cur][(j * 16) * 64] + 32, 64);  // d 32..63
            sc[j].v = wmma_bf16(qa0, kb0, z);
            sc[j].v = wmma_bf16(qa1, kb1, sc[j].v);
        }
        float mk[4];
        #pragma unroll
        for (int j = 0; j < 4; j++) mk[j] = mb[k0 + j * 16 + (lane & 15)];

        // ---- online softmax (row M = r + 8*(lane>>4), N across 16-lane group) ----
        #pragma unroll
        for (int r = 0; r < 8; r++) {
            float vmax = -1e30f;
            #pragma unroll
            for (int j = 0; j < 4; j++) {
                sc[j].f[r] = sc[j].f[r] * scale + mk[j];
                vmax = fmaxf(vmax, sc[j].f[r]);
            }
            vmax = rmax16(vmax);
            float newm  = fmaxf(mrow[r], vmax);
            float alpha = __expf(mrow[r] - newm);
            float ls = 0.0f;
            int M = r + 8 * (lane >> 4);
            #pragma unroll
            for (int j = 0; j < 4; j++) {
                float p = __expf(sc[j].f[r] - newm);
                ls += p;
                pw[M * 64 + j * 16 + (lane & 15)] = (bf16)p;
            }
            lrow[r] = lrow[r] * alpha + rsum16(ls);
            mrow[r] = newm;
            #pragma unroll
            for (int t = 0; t < 4; t++) acc[t].f[r] *= alpha;
        }

        // ---- ctx += P V ----
        v16bf pa0 = load_fragA(pw, 64);        // keys 0..31
        v16bf pa1 = load_fragA(pw + 32, 64);   // keys 32..63
        #pragma unroll
        for (int t = 0; t < 4; t++) {
            v16bf vb0 = load_fragB(&Vbuf[cur][(t * 16) * 64], 64);       // keys 0..31
            v16bf vb1 = load_fragB(&Vbuf[cur][(t * 16) * 64] + 32, 64);  // keys 32..63
            acc[t].v = wmma_bf16(pa0, vb0, acc[t].v);
            acc[t].v = wmma_bf16(pa1, vb1, acc[t].v);
        }
    }

    // ---- epilogue: ctx / l, write [b, s, h*64 + d] fp32 ----
    #pragma unroll
    for (int t = 0; t < 4; t++) {
        #pragma unroll
        for (int r = 0; r < 8; r++) {
            int q = q0 + r + 8 * (lane >> 4);
            int d = t * 16 + (lane & 15);
            out[((size_t)(b * SS + q)) * DM + h * DH + d] = acc[t].f[r] / lrow[r];
        }
    }
}

extern "C" void kernel_launch(void* const* d_in, const int* in_sizes, int n_in,
                              void* d_out, int out_size, void* d_ws, size_t ws_size,
                              hipStream_t stream) {
    const float* X    = (const float*)d_in[0];
    const float* mask = (const float*)d_in[1];
    const float* Wq   = (const float*)d_in[2];
    const float* bq   = (const float*)d_in[3];
    const float* Wk   = (const float*)d_in[4];
    const float* bk   = (const float*)d_in[5];
    const float* Wv   = (const float*)d_in[6];
    const float* bv   = (const float*)d_in[7];

    char* ws = (char*)d_ws;
    const size_t XBF_OFF = 0;                                   // 16 MB
    const size_t WT_OFF  = XBF_OFF + (size_t)BB*SS*DM*2;        //  6 MB
    const size_t Q_OFF   = WT_OFF  + (size_t)3*DM*DM*2;         // 16 MB
    const size_t K_OFF   = Q_OFF   + (size_t)BB*SS*DM*2;        // 16 MB
    const size_t VT_OFF  = K_OFF   + (size_t)BB*SS*DM*2;        // 16 MB

    bf16* Xbf = (bf16*)(ws + XBF_OFF);
    bf16* Wt  = (bf16*)(ws + WT_OFF);
    bf16* Qb  = (bf16*)(ws + Q_OFF);
    bf16* Kb  = (bf16*)(ws + K_OFF);
    bf16* Vtb = (bf16*)(ws + VT_OFF);

    cvt_x_kernel<<<dim3((BB * SS * DM) / (256 * 4)), dim3(256), 0, stream>>>(X, Xbf);
    wt_kernel<<<dim3((DM * DM) / 256, 1, 3), dim3(256), 0, stream>>>(Wq, Wk, Wv, Wt);
    qkv_gemm_kernel<<<dim3((BB * SS) / 64, DM / 64, 3), dim3(128), 0, stream>>>(
        Xbf, Wt, bq, bk, bv, Qb, Kb, Vtb);
    attn_kernel<<<dim3(SS / 64, HH, BB), dim3(128), 0, stream>>>(Qb, Kb, Vtb, mask, (float*)d_out);
}
